// MPNNLayer_83279415869562
// MI455X (gfx1250) — matrix operand order repaired and checked by reading, not compiled, not run on hardware
//
#include <hip/hip_runtime.h>
#include <hip/hip_bf16.h>

// ---------------------------------------------------------------------------
// MPNN layer for MI455X (gfx1250, wave32, WMMA).
// Edge MLPs: 16x16x32 f16 WMMA, f32 accumulation, M=32 edge tiles so each
// weight B-fragment (one 32B load/lane from L2) feeds TWO WMMAs. Weights are
// pre-packed into contiguous-k B fragments (k = 16*(lane/16) + s); activations staged in LDS (f16)
// and gathered in the same contiguous-k order (coalesces to ds_load_b128).
// Scatter-mean uses relaxed agent-scope f32 global atomics.
// ---------------------------------------------------------------------------

typedef __attribute__((ext_vector_type(16))) _Float16    v16h;
typedef __attribute__((ext_vector_type(8)))  float       v8f;
typedef __attribute__((ext_vector_type(8)))  unsigned int v8u;

#define HD 128
#define LN_EPS 1e-5f

union HU { v8u u; v16h h; };

__device__ __forceinline__ float silu_f(float x) { return x / (1.0f + __expf(-x)); }

__device__ __forceinline__ void atomic_add_f32(float* p, float v) {
  __hip_atomic_fetch_add(p, v, __ATOMIC_RELAXED, __HIP_MEMORY_SCOPE_AGENT);
}

// A-fragment gather from an LDS f16 row-major tile [rows][ldw_halves].
// A 16x32 layout matching packed B: lane(0..15)->M=lane, lanes 16..31 same M, K+16;
// VGPR p holds K = 2p + 16*(lane>=16), consecutive K pair.
__device__ __forceinline__ v16h load_a_frag(const unsigned int* sA32, int ldw_halves,
                                            int kbase, int lane) {
  const int m = lane & 15;
  const int half = lane >> 4;
  const int row = m * (ldw_halves >> 1);
  HU r;
#pragma unroll
  for (int p = 0; p < 8; ++p) {
    const int k0 = 2 * p + half * 16 + kbase;
    r.u[p] = sA32[row + (k0 >> 1)];
  }
  return r.h;
}

__device__ __forceinline__ void zero_acc(v8f* acc) {
#pragma unroll
  for (int i = 0; i < 8; ++i)
    acc[i] = (v8f){0.f, 0.f, 0.f, 0.f, 0.f, 0.f, 0.f, 0.f};
}

// 16xK tile GEMM: A rows 0..15 from LDS, packed B fragments from global.
__device__ __forceinline__ void gemm_tile(const unsigned int* sA32, int ldw, int ksteps,
                                          const v8u* Wp, v8f* acc, int lane) {
  for (int ks = 0; ks < ksteps; ++ks) {
    v16h a = load_a_frag(sA32, ldw, ks * 32, lane);
#pragma unroll
    for (int nf = 0; nf < 8; ++nf) {
      HU b;
      b.u = Wp[(ks * 8 + nf) * 32 + lane];
      acc[nf] = __builtin_amdgcn_wmma_f32_16x16x32_f16(
          false, a, false, b.h, (short)0, acc[nf], false, false);
    }
  }
}

// 32xK tile GEMM: two A fragments (rows 0..15 and 16..31) share each B
// fragment -> 2x WMMA per weight load, half the L2 weight traffic.
__device__ __forceinline__ void gemm_tile32(const unsigned int* sA32, int ldw, int ksteps,
                                            const v8u* Wp, v8f* acc0, v8f* acc1,
                                            int lane) {
  const unsigned int* sA32hi = sA32 + 16 * (ldw >> 1);
  for (int ks = 0; ks < ksteps; ++ks) {
    v16h a0 = load_a_frag(sA32,   ldw, ks * 32, lane);
    v16h a1 = load_a_frag(sA32hi, ldw, ks * 32, lane);
#pragma unroll
    for (int nf = 0; nf < 8; ++nf) {
      HU b;
      b.u = Wp[(ks * 8 + nf) * 32 + lane];
      acc0[nf] = __builtin_amdgcn_wmma_f32_16x16x32_f16(
          false, a0, false, b.h, (short)0, acc0[nf], false, false);
      acc1[nf] = __builtin_amdgcn_wmma_f32_16x16x32_f16(
          false, a1, false, b.h, (short)0, acc1[nf], false, false);
    }
  }
}

// ---------------------------------------------------------------------------
// Weight repack: W[K][128] f32 -> f16 fragments so that fragment (ks,nf),
// lane l owns 16 contiguous halves: n = nf*16 + (l&15),
// k = ks*32 + (l>>4)*16 + t  (t = 0..15).
// ---------------------------------------------------------------------------
__global__ void pack_w(const float* __restrict__ W, _Float16* __restrict__ dst, int K) {
  const int total = (K / 32) * 8 * 32 * 16;
  int idx = blockIdx.x * blockDim.x + threadIdx.x;
  if (idx >= total) return;
  const int t  = idx & 15;
  const int l  = (idx >> 4) & 31;
  const int nf = (idx >> 9) & 7;
  const int ks = idx >> 12;
  const int k = ks * 32 + (l >> 4) * 16 + t;
  const int n = nf * 16 + (l & 15);
  dst[idx] = (_Float16)W[(size_t)k * HD + n];
}

__global__ void zero_kernel(float* __restrict__ p, size_t n) {
  size_t i = (size_t)blockIdx.x * blockDim.x + threadIdx.x;
  const size_t stride = (size_t)gridDim.x * blockDim.x;
  for (; i < n; i += stride) p[i] = 0.f;
}

// ---------------------------------------------------------------------------
// Edge kernel: one wave per 32-edge tile.
// ---------------------------------------------------------------------------
__global__ void __launch_bounds__(32)
edge_kernel(const int* __restrict__ ei, const int* __restrict__ ej, int E,
            const float* __restrict__ node_emb, const float* __restrict__ pos_hid,
            const float* __restrict__ node_pos,
            const float* __restrict__ We1, const float* __restrict__ be1,
            const v8u* __restrict__ We2p, const float* __restrict__ be2,
            const v8u* __restrict__ Wm1p, const float* __restrict__ bm1,
            const v8u* __restrict__ Wm2p, const float* __restrict__ bm2,
            const v8u* __restrict__ Wc1p, const float* __restrict__ bc1,
            const v8u* __restrict__ Wc2p, const float* __restrict__ bc2,
            float* __restrict__ msg_sum, float* __restrict__ coord_sum,
            float* __restrict__ counts) {
  __shared__ __attribute__((aligned(16))) _Float16 sA[32 * 640];  // m_in tile (f16)
  __shared__ __attribute__((aligned(16))) _Float16 sH[32 * 128];  // hidden tile (f16)
  __shared__ int   sNi[32], sNj[32];
  __shared__ float sDx[32], sDy[32], sDz[32];

  const int lane = threadIdx.x;
  const int e0 = blockIdx.x * 32;

  {
    const int e = e0 + lane;
    int ni = -1, nj = 0;
    float dx = 0.f, dy = 0.f, dz = 0.f;
    if (e < E) {
      ni = ei[e];
      nj = ej[e];
      dx = node_pos[(size_t)ni * 3 + 0] - node_pos[(size_t)nj * 3 + 0];
      dy = node_pos[(size_t)ni * 3 + 1] - node_pos[(size_t)nj * 3 + 1];
      dz = node_pos[(size_t)ni * 3 + 2] - node_pos[(size_t)nj * 3 + 2];
    }
    sNi[lane] = ni; sNj[lane] = nj;
    sDx[lane] = dx; sDy[lane] = dy; sDz[lane] = dz;
  }
  __syncthreads();

  // Gather node features into m_in blocks 0,1,3,4 (f32 -> f16), float4 loads.
  const float4* ne4 = (const float4*)node_emb;
  const float4* ph4 = (const float4*)pos_hid;
  for (int t = lane; t < 32 * 32; t += 32) {
    const int m = t >> 5, q = t & 31;
    const int ni = (sNi[m] < 0) ? 0 : sNi[m];
    const int nj = sNj[m];
    const float4 a = ne4[(size_t)ni * 32 + q];
    const float4 b = ne4[(size_t)nj * 32 + q];
    const float4 c = ph4[(size_t)ni * 32 + q];
    const float4 d = ph4[(size_t)nj * 32 + q];
    const int base = m * 640, c0 = q * 4;
    sA[base +   0 + c0 + 0] = (_Float16)a.x; sA[base +   0 + c0 + 1] = (_Float16)a.y;
    sA[base +   0 + c0 + 2] = (_Float16)a.z; sA[base +   0 + c0 + 3] = (_Float16)a.w;
    sA[base + 128 + c0 + 0] = (_Float16)b.x; sA[base + 128 + c0 + 1] = (_Float16)b.y;
    sA[base + 128 + c0 + 2] = (_Float16)b.z; sA[base + 128 + c0 + 3] = (_Float16)b.w;
    sA[base + 384 + c0 + 0] = (_Float16)c.x; sA[base + 384 + c0 + 1] = (_Float16)c.y;
    sA[base + 384 + c0 + 2] = (_Float16)c.z; sA[base + 384 + c0 + 3] = (_Float16)c.w;
    sA[base + 512 + c0 + 0] = (_Float16)d.x; sA[base + 512 + c0 + 1] = (_Float16)d.y;
    sA[base + 512 + c0 + 2] = (_Float16)d.z; sA[base + 512 + c0 + 3] = (_Float16)d.w;
  }

  // edge.l1 (3 -> 128) + SiLU, pure VALU (K=3 too small for WMMA).
  for (int t = lane; t < 32 * 128; t += 32) {
    const int m = t >> 7, n = t & 127;
    const float pre = sDx[m] * We1[n] + sDy[m] * We1[128 + n] +
                      sDz[m] * We1[256 + n] + be1[n];
    sH[t] = (_Float16)silu_f(pre);
  }
  __syncthreads();

  v8f acc0[8], acc1[8];

  // edge.l2 (128 -> 128): pos_emb -> m_in block 2 (cols 256..383).
  zero_acc(acc0); zero_acc(acc1);
  gemm_tile32((const unsigned int*)sH, 128, 4, We2p, acc0, acc1, lane);
#pragma unroll
  for (int nf = 0; nf < 8; ++nf)
#pragma unroll
    for (int r = 0; r < 8; ++r) {
      const int mm = ((lane >> 4) << 3) + r;
      const int n = (nf << 4) + (lane & 15);
      sA[mm * 640 + 256 + n]        = (_Float16)(acc0[nf][r] + be2[n]);
      sA[(mm + 16) * 640 + 256 + n] = (_Float16)(acc1[nf][r] + be2[n]);
    }
  __syncthreads();

  // ---- msg head: silu(m_in @ Wm1 + bm1) @ Wm2 + bm2, scatter-add to msg_sum[i]
  zero_acc(acc0); zero_acc(acc1);
  gemm_tile32((const unsigned int*)sA, 640, 20, Wm1p, acc0, acc1, lane);
  __syncthreads();
#pragma unroll
  for (int nf = 0; nf < 8; ++nf)
#pragma unroll
    for (int r = 0; r < 8; ++r) {
      const int mm = ((lane >> 4) << 3) + r;
      const int n = (nf << 4) + (lane & 15);
      sH[mm * 128 + n]        = (_Float16)silu_f(acc0[nf][r] + bm1[n]);
      sH[(mm + 16) * 128 + n] = (_Float16)silu_f(acc1[nf][r] + bm1[n]);
    }
  __syncthreads();
  zero_acc(acc0); zero_acc(acc1);
  gemm_tile32((const unsigned int*)sH, 128, 4, Wm2p, acc0, acc1, lane);
#pragma unroll
  for (int nf = 0; nf < 8; ++nf)
#pragma unroll
    for (int r = 0; r < 8; ++r) {
      const int mm = ((lane >> 4) << 3) + r;
      const int n = (nf << 4) + (lane & 15);
      const int node0 = sNi[mm];
      const int node1 = sNi[mm + 16];
      if (node0 >= 0)
        atomic_add_f32(&msg_sum[(size_t)node0 * HD + n], acc0[nf][r] + bm2[n]);
      if (node1 >= 0)
        atomic_add_f32(&msg_sum[(size_t)node1 * HD + n], acc1[nf][r] + bm2[n]);
    }
  __syncthreads();

  // ---- coord_msg head: same A tile, different weights, scatter to coord_sum[i]
  zero_acc(acc0); zero_acc(acc1);
  gemm_tile32((const unsigned int*)sA, 640, 20, Wc1p, acc0, acc1, lane);
  __syncthreads();
#pragma unroll
  for (int nf = 0; nf < 8; ++nf)
#pragma unroll
    for (int r = 0; r < 8; ++r) {
      const int mm = ((lane >> 4) << 3) + r;
      const int n = (nf << 4) + (lane & 15);
      sH[mm * 128 + n]        = (_Float16)silu_f(acc0[nf][r] + bc1[n]);
      sH[(mm + 16) * 128 + n] = (_Float16)silu_f(acc1[nf][r] + bc1[n]);
    }
  __syncthreads();
  zero_acc(acc0); zero_acc(acc1);
  gemm_tile32((const unsigned int*)sH, 128, 4, Wc2p, acc0, acc1, lane);
#pragma unroll
  for (int nf = 0; nf < 8; ++nf)
#pragma unroll
    for (int r = 0; r < 8; ++r) {
      const int mm = ((lane >> 4) << 3) + r;
      const int n = (nf << 4) + (lane & 15);
      const int node0 = sNi[mm];
      const int node1 = sNi[mm + 16];
      if (node0 >= 0)
        atomic_add_f32(&coord_sum[(size_t)node0 * HD + n], acc0[nf][r] + bc2[n]);
      if (node1 >= 0)
        atomic_add_f32(&coord_sum[(size_t)node1 * HD + n], acc1[nf][r] + bc2[n]);
    }

  if (sNi[lane] >= 0) atomic_add_f32(&counts[sNi[lane]], 1.0f);
}

// ---------------------------------------------------------------------------
// Node kernel: one wave per 16-node tile. scatter-mean + LN + update MLPs.
// ---------------------------------------------------------------------------
__global__ void __launch_bounds__(32)
node_kernel(int N,
            const float* __restrict__ msg_sum, const float* __restrict__ coord_sum,
            const float* __restrict__ counts,
            const float* __restrict__ node_emb, const float* __restrict__ node_pos,
            const float* __restrict__ gM, const float* __restrict__ bM,
            const float* __restrict__ gC, const float* __restrict__ bC,
            const v8u* __restrict__ Wu1p, const float* __restrict__ bu1,
            const v8u* __restrict__ Wu2p, const float* __restrict__ bu2,
            const v8u* __restrict__ Wq1p, const float* __restrict__ bq1,
            const float* __restrict__ Wq2, const float* __restrict__ bq2,
            float* __restrict__ h_out, float* __restrict__ x_out) {
  __shared__ __attribute__((aligned(16))) float    sT[16 * 128];
  __shared__ __attribute__((aligned(16))) _Float16 sM[16 * 128];
  __shared__ __attribute__((aligned(16))) _Float16 sH[16 * 128];
  __shared__ float sMu[16], sRs[16];

  const int lane = threadIdx.x;
  const int n0 = blockIdx.x * 16;
  v8f acc[8];

  // ======== h path ========
  for (int t = lane; t < 2048; t += 32) {
    const int m = t >> 7, c = t & 127;
    const int node = n0 + m;
    float v = 0.f;
    if (node < N) v = msg_sum[(size_t)node * HD + c] / counts[node];
    sT[t] = v;
  }
  __syncthreads();
  if (lane < 16) {
    float mu = 0.f;
    for (int c = 0; c < 128; ++c) mu += sT[lane * 128 + c];
    mu *= (1.0f / 128.0f);
    float var = 0.f;
    for (int c = 0; c < 128; ++c) {
      const float d = sT[lane * 128 + c] - mu;
      var += d * d;
    }
    var *= (1.0f / 128.0f);
    sMu[lane] = mu;
    sRs[lane] = rsqrtf(var + LN_EPS);
  }
  __syncthreads();
  for (int t = lane; t < 2048; t += 32) {
    const int m = t >> 7, c = t & 127;
    sM[t] = (_Float16)((sT[t] - sMu[m]) * sRs[m] * gM[c] + bM[c]);
  }
  __syncthreads();

  zero_acc(acc);
  gemm_tile((const unsigned int*)sM, 128, 4, Wu1p, acc, lane);
#pragma unroll
  for (int nf = 0; nf < 8; ++nf)
#pragma unroll
    for (int r = 0; r < 8; ++r) {
      const int m = ((lane >> 4) << 3) + r;
      const int n = (nf << 4) + (lane & 15);
      sH[m * 128 + n] = (_Float16)silu_f(acc[nf][r] + bu1[n]);
    }
  __syncthreads();
  zero_acc(acc);
  gemm_tile((const unsigned int*)sH, 128, 4, Wu2p, acc, lane);
#pragma unroll
  for (int nf = 0; nf < 8; ++nf)
#pragma unroll
    for (int r = 0; r < 8; ++r) {
      const int m = ((lane >> 4) << 3) + r;
      const int n = (nf << 4) + (lane & 15);
      const int node = n0 + m;
      if (node < N)
        h_out[(size_t)node * HD + n] =
            acc[nf][r] + bu2[n] + node_emb[(size_t)node * HD + n];
    }
  __syncthreads();

  // ======== x path ========
  for (int t = lane; t < 2048; t += 32) {
    const int m = t >> 7, c = t & 127;
    const int node = n0 + m;
    float v = 0.f;
    if (node < N) v = coord_sum[(size_t)node * HD + c] / counts[node];
    sT[t] = v;
  }
  __syncthreads();
  if (lane < 16) {
    float mu = 0.f;
    for (int c = 0; c < 128; ++c) mu += sT[lane * 128 + c];
    mu *= (1.0f / 128.0f);
    float var = 0.f;
    for (int c = 0; c < 128; ++c) {
      const float d = sT[lane * 128 + c] - mu;
      var += d * d;
    }
    var *= (1.0f / 128.0f);
    sMu[lane] = mu;
    sRs[lane] = rsqrtf(var + LN_EPS);
  }
  __syncthreads();
  for (int t = lane; t < 2048; t += 32) {
    const int m = t >> 7, c = t & 127;
    sM[t] = (_Float16)((sT[t] - sMu[m]) * sRs[m] * gC[c] + bC[c]);
  }
  __syncthreads();

  zero_acc(acc);
  gemm_tile((const unsigned int*)sM, 128, 4, Wq1p, acc, lane);
  __syncthreads();
#pragma unroll
  for (int nf = 0; nf < 8; ++nf)
#pragma unroll
    for (int r = 0; r < 8; ++r) {
      const int m = ((lane >> 4) << 3) + r;
      const int n = (nf << 4) + (lane & 15);
      sT[m * 128 + n] = silu_f(acc[nf][r] + bq1[n]);  // f32 hidden for 128->3
    }
  __syncthreads();

  // coord_update.l2 (128 -> 3): tiny, VALU dot products.
  for (int t = lane; t < 48; t += 32) {
    const int m = t / 3, c = t - m * 3;
    const int node = n0 + m;
    if (node < N) {
      float s = bq2[c];
      for (int k = 0; k < 128; ++k) s += sT[m * 128 + k] * Wq2[k * 3 + c];
      x_out[(size_t)node * 3 + c] = s + node_pos[(size_t)node * 3 + c];
    }
  }
}

// ---------------------------------------------------------------------------
// Host-side orchestration.
// Input flattening assumed: setup_inputs() insertion order, params recursively
// in insertion order with leaves (w,b) / (g,b):
//   0 node_embedding, 1 pos_hidden, 2 node_pos,
//   3..6   edge.l1.{w,b}, edge.l2.{w,b}
//   7..10  msg.l1.{w,b},  msg.l2.{w,b}
//   11..14 coord_msg.l1.{w,b}, coord_msg.l2.{w,b}
//   15..18 update.l1.{w,b}, update.l2.{w,b}
//   19..22 coord_update.l1.{w,b}, coord_update.l2.{w,b}
//   23..24 msg_norm.{g,b}, 25..26 coord_norm.{g,b}, 27 edge_index[2,E]
// Output: h [N,128] then x [N,3], f32, concatenated.
// Workspace: ~0.5 MB packed weights + ~51.4 MB accumulators.
// ---------------------------------------------------------------------------
extern "C" void kernel_launch(void* const* d_in, const int* in_sizes, int n_in,
                              void* d_out, int out_size, void* d_ws, size_t ws_size,
                              hipStream_t stream) {
  const float* node_emb = (const float*)d_in[0];
  const float* pos_hid  = (const float*)d_in[1];
  const float* node_pos = (const float*)d_in[2];
  const float* We1 = (const float*)d_in[3];
  const float* be1 = (const float*)d_in[4];
  const float* We2 = (const float*)d_in[5];
  const float* be2 = (const float*)d_in[6];
  const float* Wm1 = (const float*)d_in[7];
  const float* bm1 = (const float*)d_in[8];
  const float* Wm2 = (const float*)d_in[9];
  const float* bm2 = (const float*)d_in[10];
  const float* Wc1 = (const float*)d_in[11];
  const float* bc1 = (const float*)d_in[12];
  const float* Wc2 = (const float*)d_in[13];
  const float* bc2 = (const float*)d_in[14];
  const float* Wu1 = (const float*)d_in[15];
  const float* bu1 = (const float*)d_in[16];
  const float* Wu2 = (const float*)d_in[17];
  const float* bu2 = (const float*)d_in[18];
  const float* Wq1 = (const float*)d_in[19];
  const float* bq1 = (const float*)d_in[20];
  const float* Wq2 = (const float*)d_in[21];
  const float* bq2 = (const float*)d_in[22];
  const float* gM  = (const float*)d_in[23];
  const float* bM  = (const float*)d_in[24];
  const float* gC  = (const float*)d_in[25];
  const float* bC  = (const float*)d_in[26];
  const int* eidx  = (const int*)d_in[n_in - 1];

  const int N = in_sizes[0] / HD;
  const int E = in_sizes[n_in - 1] / 2;
  const int* ei = eidx;
  const int* ej = eidx + E;

  // Workspace layout (256B aligned carve-outs).
  char* ws = (char*)d_ws;
  size_t off = 0;
  auto carve = [&](size_t bytes) -> void* {
    void* p = ws + off;
    off = (off + bytes + 255) & ~(size_t)255;
    return p;
  };
  _Float16* pWm1 = (_Float16*)carve(640 * 128 * 2);
  _Float16* pWc1 = (_Float16*)carve(640 * 128 * 2);
  _Float16* pWe2 = (_Float16*)carve(128 * 128 * 2);
  _Float16* pWm2 = (_Float16*)carve(128 * 128 * 2);
  _Float16* pWc2 = (_Float16*)carve(128 * 128 * 2);
  _Float16* pWu1 = (_Float16*)carve(128 * 128 * 2);
  _Float16* pWu2 = (_Float16*)carve(128 * 128 * 2);
  _Float16* pWq1 = (_Float16*)carve(128 * 128 * 2);
  float* msg_sum   = (float*)carve((size_t)N * HD * 4);
  float* coord_sum = (float*)carve((size_t)N * HD * 4);
  float* counts    = (float*)carve((size_t)N * 4);
  (void)ws_size; (void)out_size;

  // 1) Pack weights into WMMA B-fragment layout.
  const int thr = 256;
  auto pg = [&](int K) { return ((K / 32) * 8 * 32 * 16 + thr - 1) / thr; };
  pack_w<<<pg(640), thr, 0, stream>>>(Wm1, pWm1, 640);
  pack_w<<<pg(640), thr, 0, stream>>>(Wc1, pWc1, 640);
  pack_w<<<pg(128), thr, 0, stream>>>(We2, pWe2, 128);
  pack_w<<<pg(128), thr, 0, stream>>>(Wm2, pWm2, 128);
  pack_w<<<pg(128), thr, 0, stream>>>(Wc2, pWc2, 128);
  pack_w<<<pg(128), thr, 0, stream>>>(Wu1, pWu1, 128);
  pack_w<<<pg(128), thr, 0, stream>>>(Wu2, pWu2, 128);
  pack_w<<<pg(128), thr, 0, stream>>>(Wq1, pWq1, 128);

  // 2) Zero scatter accumulators (msg_sum, coord_sum, counts are contiguous).
  const size_t zcount =
      ((size_t)((char*)(counts + N) - (char*)msg_sum)) / sizeof(float);
  zero_kernel<<<4096, 256, 0, stream>>>(msg_sum, zcount);

  // 3) Edge pass (WMMA-heavy, M=32 tiles) with atomic scatter-add.
  edge_kernel<<<(E + 31) / 32, 32, 0, stream>>>(
      ei, ej, E, node_emb, pos_hid, node_pos, We1, be1,
      (const v8u*)pWe2, be2, (const v8u*)pWm1, bm1, (const v8u*)pWm2, bm2,
      (const v8u*)pWc1, bc1, (const v8u*)pWc2, bc2,
      msg_sum, coord_sum, counts);

  // 4) Node pass: mean + LayerNorm + update MLPs + residuals.
  float* h_out = (float*)d_out;
  float* x_out = h_out + (size_t)N * HD;
  node_kernel<<<(N + 15) / 16, 32, 0, stream>>>(
      N, msg_sum, coord_sum, counts, node_emb, node_pos,
      gM, bM, gC, bC,
      (const v8u*)pWu1, bu1, (const v8u*)pWu2, bu2,
      (const v8u*)pWq1, bq1, Wq2, bq2,
      h_out, x_out);
}
